// EntangledInterferenceLayer_56358560858679
// MI455X (gfx1250) — matrix operand order repaired and checked by reading, not compiled, not run on hardware
//
#include <hip/hip_runtime.h>

#define B_ 4
#define S_ 1024
#define D_ 1024
#define H_ 16
#define HD_ 64

typedef __bf16 bf16;
typedef bf16 v16bf __attribute__((ext_vector_type(16)));
typedef float v8f __attribute__((ext_vector_type(8)));
typedef unsigned short u16;
typedef u16 v16u __attribute__((ext_vector_type(16)));
typedef u16 v8u __attribute__((ext_vector_type(8)));
typedef u16 v4u __attribute__((ext_vector_type(4)));
typedef unsigned int v4ui __attribute__((ext_vector_type(4)));
typedef int v8i __attribute__((ext_vector_type(8)));
typedef int v4i __attribute__((ext_vector_type(4)));

__device__ __forceinline__ u16 f2bf(float f) {
    union { bf16 b; u16 u; } c;
    c.b = (bf16)f;               // hardware v_cvt_*_bf16_f32 path
    return c.u;
}

__device__ __forceinline__ v16bf as_bf16(v16u u) {
    union { v16u u; v16bf b; } c; c.u = u; return c.b;
}

// Build a 16-element bf16 fragment from two contiguous 8-element LDS reads.
__device__ __forceinline__ v16bf lds_frag(const u16* p0, const u16* p1) {
    v8u a = *(const v8u*)p0;
    v8u b = *(const v8u*)p1;
    v16u r;
#pragma unroll
    for (int i = 0; i < 8; ++i) { r[i] = a[i]; r[i + 8] = b[i]; }
    return as_bf16(r);
}

__device__ __forceinline__ v16bf neg_bf16(v16bf x) {
    union { v16u u; v16bf b; } c; c.b = x;
    c.u = c.u ^ (u16)0x8000u;
    return c.b;
}

__device__ __forceinline__ v8f wmma_bf16(v16bf a, v16bf b, v8f c) {
    return __builtin_amdgcn_wmma_f32_16x16x32_bf16(false, a, false, b, (short)0, c,
                                                   false, false);
}

__device__ __forceinline__ v8f zero8() {
    v8f z = {0.f, 0.f, 0.f, 0.f, 0.f, 0.f, 0.f, 0.f};
    return z;
}

// LDS aperture: generic address low 32 bits == LDS byte offset.
__device__ __forceinline__ unsigned lds_off(const void* p) {
    return (unsigned)(uintptr_t)p;
}

// ---------------------------------------------------------------------------
// TDM: 2-D tensor_load_to_lds. Descriptor packed per CDNA5 ISA ch.8 (D# groups).
// data_size = 2 bytes (bf16). tile_d0 elements contiguous per row; rows strided
// by stride_d0 elements in global; rows packed back-to-back in LDS.
// ---------------------------------------------------------------------------
__device__ __forceinline__ void tdm_load_2d_bf16(unsigned lds_addr, const void* gtile,
                                                 unsigned tile_d0, unsigned tile_d1,
                                                 unsigned stride_d0) {
    unsigned long long ga = (unsigned long long)(uintptr_t)gtile;
    v4ui g0;
    g0[0] = 1u;                                        // count=1 (valid descriptor)
    g0[1] = lds_addr;                                  // D#.lds_addr
    g0[2] = (unsigned)(ga & 0xffffffffu);              // global_addr[31:0]
    g0[3] = (unsigned)((ga >> 32) & 0x01ffffffu) | (2u << 30); // addr[56:32] | type=2
    v8i g1;
    g1[0] = (int)(1u << 16);                           // data_size=1 -> 2 bytes
    g1[1] = (int)((tile_d0 & 0xffffu) << 16);          // tensor_dim0[15:0]
    g1[2] = (int)(((tile_d0 >> 16) & 0xffffu) |        // tensor_dim0[31:16]
                  ((tile_d1 & 0xffffu) << 16));        // tensor_dim1[15:0]
    g1[3] = (int)(((tile_d1 >> 16) & 0xffffu) |        // tensor_dim1[31:16]
                  ((tile_d0 & 0xffffu) << 16));        // tile_dim0
    g1[4] = (int)(tile_d1 & 0xffffu);                  // tile_dim1 (tile_dim2 = 0)
    g1[5] = (int)stride_d0;                            // tensor_dim0_stride[31:0]
    g1[6] = 0;
    g1[7] = 0;
    v4i gz = {0, 0, 0, 0};
#if __clang_major__ >= 23
    v8i gz8 = {0, 0, 0, 0, 0, 0, 0, 0};
    __builtin_amdgcn_tensor_load_to_lds(g0, g1, gz, gz, gz8, 0);
#else
    __builtin_amdgcn_tensor_load_to_lds(g0, g1, gz, gz, 0);
#endif
}

// ---------------------------------------------------------------------------
// GEMM: Y[M,N] = X[M,K] @ W[K,N] + bias[N]   (f32 in/out, bf16 WMMA compute)
// Block: 256 threads = 8 waves; tile 128x128; wave tile 64x32 (4x2 WMMA tiles)
// ---------------------------------------------------------------------------
__global__ __launch_bounds__(256) void gemm_bias_kernel(
    const float* __restrict__ X, const float* __restrict__ W,
    const float* __restrict__ bias, float* __restrict__ Y,
    int M, int N, int K)
{
    __shared__ u16 aT[128][32];   // [m][k]
    __shared__ u16 bT[128][32];   // [n][k]  (K-contiguous for B fragments)

    const int tid  = threadIdx.x;
    const int lane = tid & 31;
    const int wid  = tid >> 5;
    const int l16  = lane & 15;
    const int half = lane >> 4;
    const int wm   = wid & 1;    // wave row (0..1): 64 rows each
    const int wn   = wid >> 1;   // wave col (0..3): 32 cols each
    const int m0   = blockIdx.y * 128;
    const int n0   = blockIdx.x * 128;

    v8f acc[4][2];
#pragma unroll
    for (int i = 0; i < 4; ++i)
#pragma unroll
        for (int j = 0; j < 2; ++j) acc[i][j] = zero8();

    for (int kc = 0; kc < K; kc += 32) {
        // Batch all global loads first (lets the compiler stage s_wait_loadcnt).
        float4 va[4], vb[4];
#pragma unroll
        for (int it = 0; it < 4; ++it) {
            int f = tid + 256 * it;          // 0..1023 float4 slots (A: 128x8)
            int row = f >> 3, c4 = f & 7;
            va[it] = *(const float4*)(X + (size_t)(m0 + row) * K + kc + c4 * 4);
        }
#pragma unroll
        for (int it = 0; it < 4; ++it) {
            int f = tid + 256 * it;          // B: 32x32 float4 slots
            int k = f >> 5, c4 = f & 31;
            vb[it] = *(const float4*)(W + (size_t)(kc + k) * N + n0 + c4 * 4);
        }
        if (kc + 32 < K) {  // gfx1250 global_prefetch_b8 for next K tile
            __builtin_prefetch(X + (size_t)(m0 + (tid >> 1)) * K + kc + 32, 0, 0);
            __builtin_prefetch(W + (size_t)(kc + 32 + (tid & 31)) * N + n0, 0, 0);
        }
        // Convert + stage to LDS.
#pragma unroll
        for (int it = 0; it < 4; ++it) {
            int f = tid + 256 * it;
            int row = f >> 3, c4 = f & 7;
            v4u p; p[0] = f2bf(va[it].x); p[1] = f2bf(va[it].y);
                   p[2] = f2bf(va[it].z); p[3] = f2bf(va[it].w);
            *(v4u*)&aT[row][c4 * 4] = p;
        }
#pragma unroll
        for (int it = 0; it < 4; ++it) {
            int f = tid + 256 * it;
            int k = f >> 5, c4 = f & 31;
            bT[c4 * 4 + 0][k] = f2bf(vb[it].x);
            bT[c4 * 4 + 1][k] = f2bf(vb[it].y);
            bT[c4 * 4 + 2][k] = f2bf(vb[it].z);
            bT[c4 * 4 + 3][k] = f2bf(vb[it].w);
        }
        __syncthreads();

        v16bf afr[4];
#pragma unroll
        for (int mt = 0; mt < 4; ++mt) {
            int r = wm * 64 + mt * 16 + l16;
            afr[mt] = lds_frag(&aT[r][half ? 8 : 0], &aT[r][half ? 24 : 16]);
        }
#pragma unroll
        for (int nt = 0; nt < 2; ++nt) {
            int ccol = wn * 32 + nt * 16 + l16;
            v16bf bfr = lds_frag(&bT[ccol][half * 8], &bT[ccol][half * 8 + 16]);
#pragma unroll
            for (int mt = 0; mt < 4; ++mt)
                acc[mt][nt] = wmma_bf16(afr[mt], bfr, acc[mt][nt]);
        }
        __syncthreads();
    }

#pragma unroll
    for (int mt = 0; mt < 4; ++mt)
#pragma unroll
        for (int nt = 0; nt < 2; ++nt) {
            int col = n0 + wn * 32 + nt * 16 + l16;
            float bv = bias[col];
#pragma unroll
            for (int j = 0; j < 8; ++j) {
                int row = m0 + wm * 64 + mt * 16 + (half ? j + 8 : j);
                Y[(size_t)row * N + col] = acc[mt][nt][j] + bv;
            }
        }
}

// ---------------------------------------------------------------------------
// RoPE (first 32 dims/head) + 16x16 head entanglement + phase rotation.
// Writes bf16 in attention-friendly layout [B][H][S][HD].
// ---------------------------------------------------------------------------
__global__ __launch_bounds__(256) void rope_entangle_phase_kernel(
    const float* __restrict__ inR, const float* __restrict__ inI,
    const float* __restrict__ freqs, const float* __restrict__ ent,
    const float* __restrict__ phase,
    u16* __restrict__ outR, u16* __restrict__ outI)
{
    __shared__ float lr[D_], li[D_];
    const int bs = blockIdx.x;
    const int b  = bs >> 10;            // S_ = 1024
    const int s  = bs & (S_ - 1);
    const float pos = (float)s;
    const int t = threadIdx.x;
    const int e = t * 4;
    const int dh = e & (HD_ - 1);

    float4 r4 = *(const float4*)(inR + (size_t)bs * D_ + e);
    float4 i4 = *(const float4*)(inI + (size_t)bs * D_ + e);
    float rr[4] = {r4.x, r4.y, r4.z, r4.w};
    float ii[4] = {i4.x, i4.y, i4.z, i4.w};
    if (dh < 32) {
#pragma unroll
        for (int p = 0; p < 2; ++p) {
            int d0 = dh + p * 2;
            float em = pos * freqs[d0 >> 1];
            float ce = cosf(em), se = sinf(em);
            float x0 = rr[p * 2], x1 = rr[p * 2 + 1];
            rr[p * 2]     = x0 * ce - x1 * se;
            rr[p * 2 + 1] = x1 * ce + x0 * se;
            float y0 = ii[p * 2], y1 = ii[p * 2 + 1];
            ii[p * 2]     = y0 * ce - y1 * se;
            ii[p * 2 + 1] = y1 * ce + y0 * se;
        }
    }
#pragma unroll
    for (int j = 0; j < 4; ++j) { lr[e + j] = rr[j]; li[e + j] = ii[j]; }
    __syncthreads();

    const int x = e >> 6;      // output head
    const int d = e & 63;      // head dim base (aligned to 4)
    float aR[4] = {0, 0, 0, 0}, aI[4] = {0, 0, 0, 0};
#pragma unroll
    for (int h = 0; h < H_; ++h) {
        float w = ent[h * H_ + x];
#pragma unroll
        for (int j = 0; j < 4; ++j) {
            aR[j] += lr[h * HD_ + d + j] * w;
            aI[j] += li[h * HD_ + d + j] * w;
        }
    }
    v4u pr, pi;
#pragma unroll
    for (int j = 0; j < 4; ++j) {
        float ph = phase[x * HD_ + d + j];
        float cp = cosf(ph), sp = sinf(ph);
        pr[j] = f2bf(aR[j] * cp - aI[j] * sp);
        pi[j] = f2bf(aR[j] * sp + aI[j] * cp);
    }
    size_t o = (((size_t)(b * H_ + x)) * S_ + s) * HD_ + d;
    *(v4u*)(outR + o) = pr;
    *(v4u*)(outI + o) = pi;
}

// ---------------------------------------------------------------------------
// V: f32 [B][S][D] -> bf16 transposed [B][H][HD][S] (rows contiguous per hd).
// ---------------------------------------------------------------------------
__global__ __launch_bounds__(256) void conv_v_kernel(
    const float* __restrict__ vR, const float* __restrict__ vI,
    u16* __restrict__ outR, u16* __restrict__ outI)
{
    const int bs = blockIdx.x;
    const int b  = bs >> 10;
    const int s  = bs & (S_ - 1);
    const int e  = threadIdx.x * 4;
    const int h  = e >> 6;
    const int d  = e & 63;
    float4 a = *(const float4*)(vR + (size_t)bs * D_ + e);
    float4 c = *(const float4*)(vI + (size_t)bs * D_ + e);
    float fa[4] = {a.x, a.y, a.z, a.w};
    float fc[4] = {c.x, c.y, c.z, c.w};
#pragma unroll
    for (int j = 0; j < 4; ++j) {
        size_t o = (((size_t)(b * H_ + h)) * HD_ + d + j) * S_ + s;
        outR[o] = f2bf(fa[j]);
        outI[o] = f2bf(fc[j]);
    }
}

// ---------------------------------------------------------------------------
// Flash-style complex attention. Grid (S/64, B*H). 128 threads = 4 waves,
// each wave owns 16 query rows. K/V tiles are DMA'd into LDS by the TDM
// (tensor_load_to_lds, TENSORcnt) -- one descriptor per wave per chunk.
// ---------------------------------------------------------------------------
__global__ __launch_bounds__(128) void attn_kernel(
    const u16* __restrict__ qbr, const u16* __restrict__ qbi,   // [B][H][S][HD] bf16
    const u16* __restrict__ kbr, const u16* __restrict__ kbi,   // [B][H][S][HD] bf16
    const u16* __restrict__ vtr, const u16* __restrict__ vti,   // [B][H][HD][S] bf16
    const unsigned char* __restrict__ amask,
    const float* __restrict__ strengthp, const float* __restrict__ tempp,
    float* __restrict__ oR, float* __restrict__ oI)
{
    __shared__ u16 qR[64][64], qI[64][64];     // [row][hd]
    __shared__ u16 kR[64][64], kI[64][64];     // [key][hd]
    __shared__ u16 vRT[64][64], vIT[64][64];   // [hd][key]
    __shared__ u16 pL[4][16][64];              // per-wave P tile [row][key]

    const int tid  = threadIdx.x;
    const int lane = tid & 31;
    const int wid  = tid >> 5;
    const int l16  = lane & 15;
    const int half = lane >> 4;

    const int qb = blockIdx.x;
    const int bh = blockIdx.y;
    const int b  = bh >> 4;
    const int h  = bh & 15;

    const size_t headQK = ((size_t)bh) * S_ * HD_;   // element base, [B][H][S][HD]
    const size_t headV  = ((size_t)bh) * HD_ * S_;   // element base, [B][H][HD][S]

    // DMA the 64x64 Q tiles (contiguous 8KB each) into LDS via the TDM.
    if (wid == 0) tdm_load_2d_bf16(lds_off(&qR[0][0]), qbr + headQK + (size_t)qb * 64 * HD_,
                                   64, 64, HD_);
    if (wid == 1) tdm_load_2d_bf16(lds_off(&qI[0][0]), qbi + headQK + (size_t)qb * 64 * HD_,
                                   64, 64, HD_);

    const float strength = 1.0f / (1.0f + __expf(-strengthp[0]));
    const float temp = fmaxf(tempp[0], 0.01f);
    const float smul = strength / temp;
    const float scale = 0.125f;   // 1/sqrt(64)

    v8f accR[4], accI[4];
#pragma unroll
    for (int i = 0; i < 4; ++i) { accR[i] = zero8(); accI[i] = zero8(); }
    v8f mrow, lrow;
#pragma unroll
    for (int j = 0; j < 8; ++j) { mrow[j] = -1e30f; lrow[j] = 0.f; }

    for (int kc = 0; kc <= qb; ++kc) {
        __syncthreads();   // prior chunk's LDS reads complete before DMA overwrite

        // One wave issues each K/V tile DMA (uniform branch; EXEC stays full).
        if (wid == 0) tdm_load_2d_bf16(lds_off(&kR[0][0]),
                                       kbr + headQK + (size_t)kc * 64 * HD_, 64, 64, HD_);
        else if (wid == 1) tdm_load_2d_bf16(lds_off(&kI[0][0]),
                                       kbi + headQK + (size_t)kc * 64 * HD_, 64, 64, HD_);
        else if (wid == 2) tdm_load_2d_bf16(lds_off(&vRT[0][0]),
                                       vtr + headV + (size_t)kc * 64, 64, 64, S_);
        else               tdm_load_2d_bf16(lds_off(&vIT[0][0]),
                                       vti + headV + (size_t)kc * 64, 64, 64, S_);
        __builtin_amdgcn_s_wait_tensorcnt(0);   // issuing wave: data landed in LDS
        __syncthreads();                        // visible to all waves

        // Q fragments for this wave's 16 rows (2 K-chunks of 32)
        v16bf aQR[2], aQI[2], aQRn[2];
#pragma unroll
        for (int c = 0; c < 2; ++c) {
            int r  = wid * 16 + l16;
            int o0 = c * 32 + (half ? 8 : 0);
            int o1 = c * 32 + (half ? 24 : 16);
            aQR[c]  = lds_frag(&qR[r][o0], &qR[r][o1]);
            aQI[c]  = lds_frag(&qI[r][o0], &qI[r][o1]);
            aQRn[c] = neg_bf16(aQR[c]);
        }

        v8f logit[4];
#pragma unroll
        for (int nt = 0; nt < 4; ++nt) {
            v8f cr = zero8(), ci = zero8();
#pragma unroll
            for (int c = 0; c < 2; ++c) {
                int key = nt * 16 + l16;
                int ko  = c * 32 + half * 8;
                v16bf bKR = lds_frag(&kR[key][ko], &kR[key][ko + 16]);
                v16bf bKI = lds_frag(&kI[key][ko], &kI[key][ko + 16]);
                cr = wmma_bf16(aQR[c],  bKR, cr);   // qr.kr
                cr = wmma_bf16(aQI[c],  bKI, cr);   // + qi.ki
                ci = wmma_bf16(aQI[c],  bKR, ci);   // qi.kr
                ci = wmma_bf16(aQRn[c], bKI, ci);   // - qr.ki
            }
            int col = kc * 64 + nt * 16 + l16;
            bool mcol = (amask[b * S_ + col] != 0);
#pragma unroll
            for (int j = 0; j < 8; ++j) {
                int row = qb * 64 + wid * 16 + (half ? j + 8 : j);
                float r_ = cr[j] * scale, i_ = ci[j] * scale;
                float mag = sqrtf(r_ * r_ + i_ * i_ + 1e-6f);
                float lg = mag * smul;
                if (col > row || mcol) lg = -1e30f;
                logit[nt][j] = lg;
            }
        }

        // Row max (tiles, then 16-lane butterfly within each half)
        v8f tmax = logit[0];
#pragma unroll
        for (int nt = 1; nt < 4; ++nt)
#pragma unroll
            for (int j = 0; j < 8; ++j) tmax[j] = fmaxf(tmax[j], logit[nt][j]);
#pragma unroll
        for (int off = 1; off < 16; off <<= 1)
#pragma unroll
            for (int j = 0; j < 8; ++j)
                tmax[j] = fmaxf(tmax[j], __shfl_xor(tmax[j], off, 32));

        v8f mnew, alpha;
#pragma unroll
        for (int j = 0; j < 8; ++j) {
            mnew[j]  = fmaxf(mrow[j], tmax[j]);
            alpha[j] = __expf(mrow[j] - mnew[j]);
        }

        v8f psum = zero8();
#pragma unroll
        for (int nt = 0; nt < 4; ++nt)
#pragma unroll
            for (int j = 0; j < 8; ++j) {
                float p = __expf(logit[nt][j] - mnew[j]);
                psum[j] += p;
                pL[wid][half ? j + 8 : j][nt * 16 + l16] = f2bf(p);
            }
#pragma unroll
        for (int off = 1; off < 16; off <<= 1)
#pragma unroll
            for (int j = 0; j < 8; ++j) psum[j] += __shfl_xor(psum[j], off, 32);

#pragma unroll
        for (int j = 0; j < 8; ++j) {
            lrow[j] = lrow[j] * alpha[j] + psum[j];
            mrow[j] = mnew[j];
        }
#pragma unroll
        for (int nt = 0; nt < 4; ++nt)
#pragma unroll
            for (int j = 0; j < 8; ++j) {
                accR[nt][j] *= alpha[j];
                accI[nt][j] *= alpha[j];
            }

        // P x V (per-wave P, transposed V so B fragments are K-contiguous)
#pragma unroll
        for (int c = 0; c < 2; ++c) {
            int o0 = c * 32 + (half ? 8 : 0);
            int o1 = c * 32 + (half ? 24 : 16);
            v16bf aP = lds_frag(&pL[wid][l16][o0], &pL[wid][l16][o1]);
#pragma unroll
            for (int nt = 0; nt < 4; ++nt) {
                int hd = nt * 16 + l16;
                int ko = c * 32 + half * 8;
                v16bf bVR = lds_frag(&vRT[hd][ko], &vRT[hd][ko + 16]);
                v16bf bVI = lds_frag(&vIT[hd][ko], &vIT[hd][ko + 16]);
                accR[nt] = wmma_bf16(aP, bVR, accR[nt]);
                accI[nt] = wmma_bf16(aP, bVI, accI[nt]);
            }
        }
    }

    // Finalize: divide by softmax denominator, write f32 [b][s][h*64+hd]
#pragma unroll
    for (int nt = 0; nt < 4; ++nt)
#pragma unroll
        for (int j = 0; j < 8; ++j) {
            int row = qb * 64 + wid * 16 + (half ? j + 8 : j);
            int col = nt * 16 + l16;
            float inv = 1.0f / lrow[j];
            size_t idx = ((size_t)(b * S_ + row)) * D_ + h * HD_ + col;
            oR[idx] = accR[nt][j] * inv;
            oI[idx] = accI[nt][j] * inv;
        }
}

// ---------------------------------------------------------------------------
extern "C" void kernel_launch(void* const* d_in, const int* in_sizes, int n_in,
                              void* d_out, int out_size, void* d_ws, size_t ws_size,
                              hipStream_t stream)
{
    (void)in_sizes; (void)n_in; (void)out_size; (void)ws_size;

    const float* real  = (const float*)d_in[0];
    const float* imag  = (const float*)d_in[1];
    const unsigned char* amask = (const unsigned char*)d_in[2];
    const float* wq_r = (const float*)d_in[3];
    const float* bq_r = (const float*)d_in[4];
    const float* wk_r = (const float*)d_in[5];
    const float* bk_r = (const float*)d_in[6];
    const float* wv_r = (const float*)d_in[7];
    const float* bv_r = (const float*)d_in[8];
    const float* wq_i = (const float*)d_in[9];
    const float* bq_i = (const float*)d_in[10];
    const float* wk_i = (const float*)d_in[11];
    const float* bk_i = (const float*)d_in[12];
    const float* wv_i = (const float*)d_in[13];
    const float* bv_i = (const float*)d_in[14];
    const float* wo_r = (const float*)d_in[15];
    const float* bo_r = (const float*)d_in[16];
    const float* wo_i = (const float*)d_in[17];
    const float* bo_i = (const float*)d_in[18];
    const float* phase = (const float*)d_in[19];
    const float* ent   = (const float*)d_in[20];
    const float* freqs = (const float*)d_in[21];
    const float* strength = (const float*)d_in[22];
    const float* temp     = (const float*)d_in[23];

    float* ws = (float*)d_ws;
    const size_t NB = (size_t)B_ * S_ * D_;
    // f32 buffers
    float* q_r = ws + 0 * NB;  float* k_r = ws + 1 * NB;  float* v_r = ws + 2 * NB;
    float* q_i = ws + 3 * NB;  float* k_i = ws + 4 * NB;  float* v_i = ws + 5 * NB;
    float* o_r = ws + 6 * NB;  float* o_i = ws + 7 * NB;
    // bf16 buffers (attention operands)
    u16* bb  = (u16*)(ws + 8 * NB);
    u16* qbr = bb + 0 * NB;  u16* qbi = bb + 1 * NB;
    u16* kbr = bb + 2 * NB;  u16* kbi = bb + 3 * NB;
    u16* vtr = bb + 4 * NB;  u16* vti = bb + 5 * NB;

    const int M = B_ * S_;
    dim3 gg(D_ / 128, M / 128);   // (8, 32)
    dim3 gb(256);

    gemm_bias_kernel<<<gg, gb, 0, stream>>>(real, wq_r, bq_r, q_r, M, D_, D_);
    gemm_bias_kernel<<<gg, gb, 0, stream>>>(real, wk_r, bk_r, k_r, M, D_, D_);
    gemm_bias_kernel<<<gg, gb, 0, stream>>>(real, wv_r, bv_r, v_r, M, D_, D_);
    gemm_bias_kernel<<<gg, gb, 0, stream>>>(imag, wq_i, bq_i, q_i, M, D_, D_);
    gemm_bias_kernel<<<gg, gb, 0, stream>>>(imag, wk_i, bk_i, k_i, M, D_, D_);
    gemm_bias_kernel<<<gg, gb, 0, stream>>>(imag, wv_i, bv_i, v_i, M, D_, D_);

    rope_entangle_phase_kernel<<<dim3(B_ * S_), dim3(256), 0, stream>>>(
        q_r, q_i, freqs, ent, phase, qbr, qbi);
    rope_entangle_phase_kernel<<<dim3(B_ * S_), dim3(256), 0, stream>>>(
        k_r, k_i, freqs, ent, phase, kbr, kbi);
    conv_v_kernel<<<dim3(B_ * S_), dim3(256), 0, stream>>>(v_r, v_i, vtr, vti);

    attn_kernel<<<dim3(S_ / 64, B_ * H_), dim3(128), 0, stream>>>(
        qbr, qbi, kbr, kbi, vtr, vti, amask, strength, temp, o_r, o_i);

    gemm_bias_kernel<<<gg, gb, 0, stream>>>(o_r, wo_r, bo_r, (float*)d_out, M, D_, D_);
    gemm_bias_kernel<<<gg, gb, 0, stream>>>(o_i, wo_i, bo_i, (float*)d_out + NB, M, D_, D_);
}
